// PaiConv_43559558316207
// MI455X (gfx1250) — matrix operand order repaired and checked by reading, not compiled, not run on hardware
//
#include <hip/hip_runtime.h>

#define BDIM   128
#define WAVES  4
#define NPTS   20000
#define BATCH  8
#define KS     12          // spiral size
#define FEAT   64
#define OUTF   128
#define DTOT   (KS * FEAT) // 768 = flattened dim, 24 k-steps of 32
#define KSTEPS (DTOT / 32)
#define AS     776         // padded LDS row stride in halves (768+8) -> conflict-free

typedef __attribute__((ext_vector_type(16))) _Float16 v16h;
typedef __attribute__((ext_vector_type(8)))  _Float16 v8h;
typedef __attribute__((ext_vector_type(8)))  float    v8f;

__device__ __forceinline__ float elu_f(float v) {
    return v > 0.0f ? v : (__expf(v) - 1.0f);
}

// One-shot conversion of W [128x768] f32 -> f16 scratch (L2-resident, reused by all waves)
__global__ void w_to_f16(const float* __restrict__ W, _Float16* __restrict__ Wh, int n) {
    int i = blockIdx.x * blockDim.x + threadIdx.x;
    if (i < n) Wh[i] = (_Float16)W[i];
}

__global__ __launch_bounds__(BDIM) void paiconv_fused(
    const float* __restrict__ x,      // [B, N, F]
    const int*   __restrict__ adj,    // [B, N, K]
    const float* __restrict__ adjw,   // [N, K, K]
    const _Float16* __restrict__ Wh,  // [OUT, D] f16
    const float* __restrict__ bias,   // [OUT]
    float*       __restrict__ out)    // [B, N, OUT]
{
    extern __shared__ _Float16 smem[];
    const int lane = threadIdx.x & 31;
    const int wave = threadIdx.x >> 5;
    const int g    = lane >> 4;   // half-wave selector for WMMA layouts
    const int lm   = lane & 15;
    _Float16* Alds = smem + wave * (16 * AS);
    const int r0 = (blockIdx.x * WAVES + wave) * 16;   // first row of this wave's tile

    // ---------------- Stage A: gather + adjweight matmul + ELU -> LDS f16 tile [16 x 768]
    for (int m = 0; m < 16; ++m) {
        const int r  = r0 + m;
        const int bi = r / NPTS;
        const int ni = r - bi * NPTS;

        int myidx = 0;
        if (lane < KS) myidx = adj[((size_t)bi * NPTS + ni) * KS + lane];

        float awbuf[5];                      // 144 adjweight values spread over lanes
#pragma unroll
        for (int j = 0; j < 5; ++j) {
            int t = j * 32 + lane;
            awbuf[j] = (t < KS * KS) ? adjw[(size_t)ni * (KS * KS) + t] : 0.0f;
        }

        float xv0[KS], xv1[KS];              // gathered neighbors: lane covers f=lane, f=lane+32
#pragma unroll
        for (int k = 0; k < KS; ++k) {
            int ik = __shfl(myidx, k, 32);
            const float* xp = x + ((size_t)bi * NPTS + ik) * FEAT;
            xv0[k] = xp[lane];
            xv1[k] = xp[lane + 32];
        }

#pragma unroll
        for (int s = 0; s < KS; ++s) {
            float a0 = 0.0f, a1 = 0.0f;
#pragma unroll
            for (int k = 0; k < KS; ++k) {
                const int lin = k * KS + s;
                float aw = __shfl(awbuf[lin >> 5], lin & 31, 32);
                a0 = fmaf(xv0[k], aw, a0);
                a1 = fmaf(xv1[k], aw, a1);
            }
            Alds[m * AS + s * FEAT + lane]      = (_Float16)elu_f(a0);
            Alds[m * AS + s * FEAT + lane + 32] = (_Float16)elu_f(a1);
        }
    }
    // Same-wave LDS write->read: compiler inserts s_wait_dscnt; no barrier needed
    // (each wave touches only its private LDS region).

    // ---------------- Stage B: D[16x128] = A[16x768] x W^T via v_wmma_f32_16x16x32_f16
    float  mrow[8];
    size_t rowoff[8];
#pragma unroll
    for (int v = 0; v < 8; ++v) {
        int r  = r0 + v + 8 * g;             // D layout: VGPR v holds M = v + 8*g
        int ni = r % NPTS;
        mrow[v]   = (ni == NPTS - 1) ? 0.0f : 1.0f;   // last-node mask
        rowoff[v] = (size_t)r * OUTF;
    }

    // A fragment: lane = row lm; halves 0-15 -> K = 32t + 16g .. +15
    const _Float16* Abase = Alds + lm * AS + 16 * g;

    for (int nt = 0; nt < 8; ++nt) {
        const int o = nt * 16 + lm;          // B/D column = lane&15
        const _Float16* wrow = Wh + (size_t)o * DTOT + 16 * g;  // B: lane holds K=32t+16g+h of col o
        v8f acc = {};
#pragma unroll
        for (int t = 0; t < KSTEPS; ++t) {
            const v8h c0 = *(const v8h*)(Abase + 32 * t);
            const v8h c1 = *(const v8h*)(Abase + 32 * t + 8);
            const v16h afrag = __builtin_shufflevector(
                c0, c1, 0,1,2,3,4,5,6,7,8,9,10,11,12,13,14,15);
            const v16h bfrag = *(const v16h*)(wrow + 32 * t);
            acc = __builtin_amdgcn_wmma_f32_16x16x32_f16(
                false, afrag, false, bfrag, (short)0, acc, false, false);
        }
        const float bv = bias[o];
#pragma unroll
        for (int v = 0; v < 8; ++v) {
            out[rowoff[v] + o] = elu_f(acc[v] + bv) * mrow[v];
        }
    }
}

extern "C" void kernel_launch(void* const* d_in, const int* in_sizes, int n_in,
                              void* d_out, int out_size, void* d_ws, size_t ws_size,
                              hipStream_t stream) {
    const float* x    = (const float*)d_in[0];   // [8, 20000, 64]
    const int*   adj  = (const int*)  d_in[1];   // [8, 20000, 12]
    // d_in[2] = kernal_weight (unused by forward)
    const float* adjw = (const float*)d_in[3];   // [20000, 12, 12]
    const float* W    = (const float*)d_in[4];   // [128, 768]
    const float* bias = (const float*)d_in[5];   // [128]
    _Float16* Wh  = (_Float16*)d_ws;             // 192 KB f16 copy of W
    float*    out = (float*)d_out;               // [8, 20000, 128]

    const int wn = OUTF * DTOT;
    w_to_f16<<<(wn + 255) / 256, 256, 0, stream>>>(W, Wh, wn);

    const int rows   = BATCH * NPTS;             // 160000
    const int blocks = rows / (WAVES * 16);      // 2500, exact
    const size_t shmem = (size_t)WAVES * 16 * AS * sizeof(_Float16);  // ~97 KB
    paiconv_fused<<<blocks, BDIM, shmem, stream>>>(x, adj, adjw, Wh, bias, out);
}
